// RAttention_31533649887705
// MI455X (gfx1250) — matrix operand order repaired and checked by reading, not compiled, not run on hardware
//
#include <hip/hip_runtime.h>
#include <hip/hip_bf16.h>

#define DEVQ __device__ __forceinline__

typedef __attribute__((ext_vector_type(16))) __bf16 v16bf;
typedef __attribute__((ext_vector_type(8)))  __bf16 v8bf;
typedef __attribute__((ext_vector_type(4)))  __bf16 v4bf;
typedef __attribute__((ext_vector_type(2)))  __bf16 v2bf;
typedef __attribute__((ext_vector_type(8)))  float  v8f;
typedef __attribute__((ext_vector_type(4)))  unsigned int v4u;
typedef __attribute__((ext_vector_type(8)))  int v8i;
typedef __attribute__((ext_vector_type(4)))  int v4i;

// Problem sizes (from setup_inputs): B=2, N=2048, C=1024, H=16, hd=64
constexpr int Bb = 2;
constexpr int Nn = 2048;
constexpr int Cc = 1024;
constexpr int Hh = 16;
constexpr int HD = 64;
constexpr int Mrows = Bb * Nn;   // 4096

// ---------------------------------------------------------------------------
// WMMA helpers. CDNA5 v_wmma_f32_16x16x32_bf16:  D(16x16 f32) = A(16x32) x B(32x16) + C
// A as loaded by load_a32 (B uses the same k order): lane half = lane>>4, row m = lane&15; comp i -> k = half*16 + i
// B as loaded by load_b32: lane half = lane>>4, col n = lane&15; comp i -> k = half*16 + i
// C/D:      lane holds col n = lane&15, comp r -> row m = r + 8*half.
// ---------------------------------------------------------------------------
DEVQ v16bf load_a32(const __bf16* p, int half) {
  const v8bf lo = *(const v8bf*)(p + half * 16);
  const v8bf hi = *(const v8bf*)(p + half * 16 + 8);
  v16bf r;
#pragma unroll
  for (int i = 0; i < 8; ++i) { r[i] = lo[i]; r[i + 8] = hi[i]; }
  return r;
}

DEVQ v16bf load_b32(const __bf16* p, int half) {
  return *(const v16bf*)(p + half * 16);
}

DEVQ v8f wmma_bf16(v16bf a, v16bf b, v8f c) {
  return __builtin_amdgcn_wmma_f32_16x16x32_bf16(false, a, false, b, (short)0, c,
                                                 false, false);
}

// ---------------------------------------------------------------------------
// Tensor Data Mover: 2-D tile DMA (global -> LDS), bf16 elements.
// D# per CDNA5 ISA 8.3/8.4: group0 = {count|flags, lds_addr, global_addr,
// type=2}; group1 = {mask/data_size, tensor dims, tile dims, dim0 stride}.
// Descriptor words forced to SGPRs (VADDR0/1 are SGPR groups).
// ---------------------------------------------------------------------------
DEVQ int rfl(unsigned v) { return __builtin_amdgcn_readfirstlane((int)v); }

DEVQ void tdm_load_2d(unsigned lds_off, const void* gptr,
                      unsigned tensor_d0, unsigned tensor_d1,
                      unsigned tile_d0, unsigned tile_d1,
                      unsigned stride0_elems) {
  const unsigned long long ga = (unsigned long long)(size_t)gptr;
  v4u g0;
  g0[0] = (unsigned)rfl(1u);                       // count=1, user mode
  g0[1] = (unsigned)rfl(lds_off);                  // LDS byte address
  g0[2] = (unsigned)rfl((unsigned)(ga & 0xffffffffu));
  g0[3] = (unsigned)rfl((unsigned)((ga >> 32) & 0x01ffffffu) | (2u << 30));
  v8i g1;
  g1[0] = rfl(1u << 16);                                       // data_size=2B
  g1[1] = rfl((tensor_d0 & 0xffffu) << 16);                    // dim0 lo
  g1[2] = rfl((tensor_d0 >> 16) | ((tensor_d1 & 0xffffu) << 16));
  g1[3] = rfl((tensor_d1 >> 16) | (tile_d0 << 16));            // tile_dim0
  g1[4] = rfl(tile_d1 & 0xffffu);                              // tile_dim1, tile_dim2=0
  g1[5] = rfl(stride0_elems);                                  // dim0 stride lo
  g1[6] = 0;
  g1[7] = 0;
  v4i z = {0, 0, 0, 0};
#if defined(__clang_major__) && (__clang_major__ >= 23)
  v8i z8 = {0, 0, 0, 0, 0, 0, 0, 0};
  __builtin_amdgcn_tensor_load_to_lds(g0, g1, z, z, z8, 0);
#else
  __builtin_amdgcn_tensor_load_to_lds(g0, g1, z, z, 0);
#endif
}

DEVQ unsigned lds_offset_of(const void* p) { return (unsigned)(size_t)p; }

// ---------------------------------------------------------------------------
// Kernel 1: fp32 -> bf16 cast (vectorized x4)
// ---------------------------------------------------------------------------
__global__ void cast_f32_to_bf16(const float* __restrict__ in,
                                 __bf16* __restrict__ out, int n4) {
  int i = blockIdx.x * blockDim.x + threadIdx.x;
  if (i < n4) {
    const float4 v = ((const float4*)in)[i];
    v4bf o;
    o[0] = (__bf16)v.x; o[1] = (__bf16)v.y; o[2] = (__bf16)v.z; o[3] = (__bf16)v.w;
    ((v4bf*)out)[i] = o;
  }
}

// ---------------------------------------------------------------------------
// Kernel 2/5: NT GEMM  C[m][n] = sum_k A[m][k] * B[n][k]  (+ bias[n])
// A: [M,K] bf16 row-major, B: [N,K] bf16 row-major, C: [M,N] fp32.
// One wave computes a 32x64 output tile: 8 WMMAs per 32-wide K step against
// 12 B128 loads (B tiles reused across two A row-tiles).
// ---------------------------------------------------------------------------
__global__ void __launch_bounds__(128)
gemm_nt_bf16(const __bf16* __restrict__ A, const __bf16* __restrict__ Bm,
             float* __restrict__ Cm, const float* __restrict__ bias,
             int M, int Ncols, int K, int tiles) {
  const int wid = blockIdx.x * (blockDim.x >> 5) + (threadIdx.x >> 5);
  if (wid >= tiles) return;
  const int lane = threadIdx.x & 31;
  const int half = lane >> 4;
  const int l16  = lane & 15;
  const int nt64 = Ncols >> 6;
  const int tm = wid / nt64;   // 32-row tiles
  const int tn = wid % nt64;   // 64-col tiles

  const __bf16* arow0 = A + (size_t)(tm * 32 + l16) * K;
  const __bf16* arow1 = arow0 + (size_t)16 * K;
  const __bf16* brow[4];
  v8f acc[2][4];
#pragma unroll
  for (int t = 0; t < 4; ++t) {
    const int n = tn * 64 + t * 16 + l16;
    brow[t] = Bm + (size_t)n * K;
    const float bv = bias ? bias[n] : 0.0f;
#pragma unroll
    for (int r = 0; r < 8; ++r) { acc[0][t][r] = bv; acc[1][t][r] = bv; }
  }

  for (int ks = 0; ks < K; ks += 32) {
    const v16bf a0 = load_a32(arow0 + ks, half);
    const v16bf a1 = load_a32(arow1 + ks, half);
#pragma unroll
    for (int t = 0; t < 4; ++t) {
      const v16bf b = load_b32(brow[t] + ks, half);
      acc[0][t] = wmma_bf16(a0, b, acc[0][t]);
      acc[1][t] = wmma_bf16(a1, b, acc[1][t]);
    }
  }

#pragma unroll
  for (int u = 0; u < 2; ++u)
#pragma unroll
    for (int t = 0; t < 4; ++t) {
      const int n = tn * 64 + t * 16 + l16;
#pragma unroll
      for (int r = 0; r < 8; ++r)
        Cm[(size_t)(tm * 32 + u * 16 + r + 8 * half) * Ncols + n] = acc[u][t][r];
    }
}

// ---------------------------------------------------------------------------
// Kernel 3: per-head RMSNorm + RoPE, fp32 -> bf16 repack.
// qkv fp32 is [B,N,3,H,64]. One wave per (b,n,h); lane handles pair (2d,2d+1).
// Writes q,k as [B,H,N,64] bf16 and v transposed as [B,H,64,N] bf16.
// ---------------------------------------------------------------------------
__global__ void __launch_bounds__(128)
prep_norm_rope(const float* __restrict__ qkv, const float* __restrict__ qw,
               const float* __restrict__ kw, const float* __restrict__ pcos,
               const float* __restrict__ psin, __bf16* __restrict__ qb,
               __bf16* __restrict__ kb, __bf16* __restrict__ vT) {
  const int wid  = blockIdx.x * (blockDim.x >> 5) + (threadIdx.x >> 5);
  const int lane = threadIdx.x & 31;
  const int h = wid % Hh;
  const int n = (wid / Hh) % Nn;
  const int b = wid / (Hh * Nn);
  if (b >= Bb) return;
  const size_t m  = (size_t)b * Nn + n;
  const int    d0 = lane * 2;

  const float* base = qkv + m * (3 * Cc) + h * HD;
  const float2 q = *(const float2*)(base + d0);
  const float2 k = *(const float2*)(base + Cc + d0);
  const float2 v = *(const float2*)(base + 2 * Cc + d0);

  float sq = q.x * q.x + q.y * q.y;
  float sk = k.x * k.x + k.y * k.y;
#pragma unroll
  for (int off = 1; off < 32; off <<= 1) {
    sq += __shfl_xor(sq, off, 32);
    sk += __shfl_xor(sk, off, 32);
  }
  const float rq = rsqrtf(sq * (1.0f / HD) + 1e-6f);
  const float rk = rsqrtf(sk * (1.0f / HD) + 1e-6f);

  const float2 wq = *(const float2*)(qw + d0);
  const float2 wk = *(const float2*)(kw + d0);
  const float qr = q.x * rq * wq.x, qi = q.y * rq * wq.y;
  const float kr = k.x * rk * wk.x, ki = k.y * rk * wk.y;

  const float c = pcos[n * (HD / 2) + lane];
  const float s = psin[n * (HD / 2) + lane];

  const size_t bh = (size_t)b * Hh + h;
  {
    v2bf o; o[0] = (__bf16)(qr * c - qi * s); o[1] = (__bf16)(qr * s + qi * c);
    *(v2bf*)(qb + (bh * Nn + n) * HD + d0) = o;
  }
  {
    v2bf o; o[0] = (__bf16)(kr * c - ki * s); o[1] = (__bf16)(kr * s + ki * c);
    *(v2bf*)(kb + (bh * Nn + n) * HD + d0) = o;
  }
  __bf16* vp = vT + (bh * HD + d0) * (size_t)Nn + n;
  vp[0]  = (__bf16)v.x;
  vp[Nn] = (__bf16)v.y;
}

// ---------------------------------------------------------------------------
// Kernel 4: causal flash attention, one wave per (b,h, 16-row q tile).
// K/V tiles are DMA'd into double-buffered LDS by the Tensor Data Mover
// (tensor_load_to_lds, TENSORcnt-synced); online softmax over 32-key chunks;
// S and P*V via WMMA bf16; P transposed C-layout -> A-layout through per-wave
// LDS staging (per-wave s_wait_dscnt, no block barriers: divergent chunk
// counts). Output bf16 rows of [M, C] for the projection GEMM.
// ---------------------------------------------------------------------------
__global__ void __launch_bounds__(64)
flash_attn(const __bf16* __restrict__ qb, const __bf16* __restrict__ kb,
           const __bf16* __restrict__ vT, __bf16* __restrict__ outb) {
  __shared__ __attribute__((aligned(64))) __bf16 kbuf[2][2][32 * 64]; // 16 KB
  __shared__ __attribute__((aligned(64))) __bf16 vbuf[2][2][64 * 32]; // 16 KB
  __shared__ __attribute__((aligned(64))) __bf16 ldsP[2][16 * 32];    //  2 KB
  const int warp = threadIdx.x >> 5;
  const int wid  = blockIdx.x * 2 + warp;
  const int lane = threadIdx.x & 31;
  const int half = lane >> 4;
  const int l16  = lane & 15;
  const int QT = Nn / 16;
  const int qt = wid % QT;
  const int bh = wid / QT;
  if (bh >= Bb * Hh) return;
  const int b = bh / Hh, h = bh % Hh;

  const __bf16* qbase = qb + ((size_t)bh * Nn + qt * 16) * HD;
  const __bf16* kbase = kb + (size_t)bh * Nn * HD;
  const __bf16* vbase = vT + (size_t)bh * HD * Nn;

  // Q tile as WMMA A operand, two 32-wide K chunks over hd=64.
  const v16bf aq0 = load_a32(qbase + l16 * HD, half);
  const v16bf aq1 = load_a32(qbase + l16 * HD + 32, half);

  v8f acc[4];
  float rowm[8], rowl[8];
#pragma unroll
  for (int t = 0; t < 4; ++t)
#pragma unroll
    for (int r = 0; r < 8; ++r) acc[t][r] = 0.0f;
#pragma unroll
  for (int r = 0; r < 8; ++r) { rowm[r] = -1e30f; rowl[r] = 0.0f; }

  const float scale = 0.125f;  // hd^-0.5, hd=64
  const int nkeys = qt * 16 + 16;
  const int nch   = (nkeys + 31) >> 5;
  __bf16* myP = &ldsP[warp][0];

  // TDM prefetch of chunk 0: K rows [0..31]x64 and V^T block 64x[0..31].
  tdm_load_2d(lds_offset_of(&kbuf[warp][0][0]), kbase, HD, Nn, HD, 32, HD);
  tdm_load_2d(lds_offset_of(&vbuf[warp][0][0]), vbase, Nn, HD, 32, HD, Nn);

  for (int c = 0; c < nch; ++c) {
    const int k0 = c * 32;
    const bool more = (c + 1) < nch;
    if (more) {
      const int kn = k0 + 32;
      tdm_load_2d(lds_offset_of(&kbuf[warp][(c + 1) & 1][0]),
                  kbase + (size_t)kn * HD, HD, Nn, HD, 32, HD);
      tdm_load_2d(lds_offset_of(&vbuf[warp][(c + 1) & 1][0]),
                  vbase + kn, Nn, HD, 32, HD, Nn);
    }
    // Current chunk's two TDM loads (issued earlier, in-order) are complete
    // once at most the two just-issued remain outstanding.
    asm volatile("" ::: "memory");
    if (more) __builtin_amdgcn_s_wait_tensorcnt(2);
    else      __builtin_amdgcn_s_wait_tensorcnt(0);
    asm volatile("" ::: "memory");

    const __bf16* kl = &kbuf[warp][c & 1][0];   // [32 keys][64 dims]
    const __bf16* vl = &vbuf[warp][c & 1][0];   // [64 dims][32 keys]

    // S = Q K^T for 32 keys (two 16x16 tiles), B operand from LDS
    v8f s[2];
#pragma unroll
    for (int j = 0; j < 2; ++j) {
      const __bf16* krow = kl + (j * 16 + l16) * HD;
      const v16bf bk0 = load_b32(krow, half);
      const v16bf bk1 = load_b32(krow + 32, half);
      v8f z;
#pragma unroll
      for (int r = 0; r < 8; ++r) z[r] = 0.0f;
      z = wmma_bf16(aq0, bk0, z);
      z = wmma_bf16(aq1, bk1, z);
      s[j] = z;
    }
    // scale + causal mask
#pragma unroll
    for (int j = 0; j < 2; ++j) {
      const int kcol = k0 + j * 16 + l16;
#pragma unroll
      for (int r = 0; r < 8; ++r) {
        const int qrow = qt * 16 + r + 8 * half;
        s[j][r] = (kcol <= qrow) ? s[j][r] * scale : -1e30f;
      }
    }
    // row max over the 32 columns (16-lane half holds one row per component)
    float mx[8];
#pragma unroll
    for (int r = 0; r < 8; ++r) mx[r] = fmaxf(s[0][r], s[1][r]);
#pragma unroll
    for (int off = 1; off < 16; off <<= 1)
#pragma unroll
      for (int r = 0; r < 8; ++r) mx[r] = fmaxf(mx[r], __shfl_xor(mx[r], off, 32));

    float alpha[8];
#pragma unroll
    for (int r = 0; r < 8; ++r) {
      const float mnew = fmaxf(rowm[r], mx[r]);
      alpha[r] = __expf(rowm[r] - mnew);
      rowm[r]  = mnew;
    }
    float psum[8];
#pragma unroll
    for (int r = 0; r < 8; ++r) psum[r] = 0.0f;
#pragma unroll
    for (int j = 0; j < 2; ++j)
#pragma unroll
      for (int r = 0; r < 8; ++r) {
        const float pv = __expf(s[j][r] - rowm[r]);
        s[j][r] = pv;
        psum[r] += pv;
      }
#pragma unroll
    for (int off = 1; off < 16; off <<= 1)
#pragma unroll
      for (int r = 0; r < 8; ++r) psum[r] += __shfl_xor(psum[r], off, 32);
#pragma unroll
    for (int r = 0; r < 8; ++r) rowl[r] = rowl[r] * alpha[r] + psum[r];
#pragma unroll
    for (int t = 0; t < 4; ++t)
#pragma unroll
      for (int r = 0; r < 8; ++r) acc[t][r] *= alpha[r];

    // P: C-layout -> LDS (row-major 16x32) -> A-layout
#pragma unroll
    for (int j = 0; j < 2; ++j)
#pragma unroll
      for (int r = 0; r < 8; ++r)
        myP[(r + 8 * half) * 32 + j * 16 + l16] = (__bf16)s[j][r];
    asm volatile("s_wait_dscnt 0" ::: "memory");
    const v16bf pa = load_a32(myP + l16 * 32, half);

    // O += P V  (V^T tile in LDS, contiguous in key dim)
#pragma unroll
    for (int t = 0; t < 4; ++t) {
      const v16bf bv = load_b32(vl + (t * 16 + l16) * 32, half);
      acc[t] = wmma_bf16(pa, bv, acc[t]);
    }
  }

  // epilogue: divide by row sums, write bf16 rows of [M, C]
#pragma unroll
  for (int r = 0; r < 8; ++r) rowl[r] = 1.0f / rowl[r];
#pragma unroll
  for (int t = 0; t < 4; ++t)
#pragma unroll
    for (int r = 0; r < 8; ++r) {
      const int qrow = qt * 16 + r + 8 * half;
      const size_t off =
          ((size_t)b * Nn + qrow) * Cc + h * HD + t * 16 + l16;
      outb[off] = (__bf16)(acc[t][r] * rowl[r]);
    }
}

// ---------------------------------------------------------------------------
// Host launcher
// ---------------------------------------------------------------------------
extern "C" void kernel_launch(void* const* d_in, const int* in_sizes, int n_in,
                              void* d_out, int out_size, void* d_ws,
                              size_t ws_size, hipStream_t stream) {
  const float* x         = (const float*)d_in[0];
  const float* qkv_w     = (const float*)d_in[1];
  const float* q_norm_w  = (const float*)d_in[2];
  const float* k_norm_w  = (const float*)d_in[3];
  const float* proj_w    = (const float*)d_in[4];
  const float* proj_b    = (const float*)d_in[5];
  const float* pos_cos   = (const float*)d_in[6];
  const float* pos_sin   = (const float*)d_in[7];
  // d_in[8] = mask: causal mask computed analytically in-kernel.
  float* out = (float*)d_out;

  char* ws = (char*)d_ws;
  size_t off = 0;
  auto carve = [&](size_t bytes) -> void* {
    void* p = ws + off;
    off = (off + bytes + 255) & ~(size_t)255;
    return p;
  };
  __bf16* xb      = (__bf16*)carve((size_t)Mrows * Cc * 2);       //  8 MB
  __bf16* wqkvb   = (__bf16*)carve((size_t)3 * Cc * Cc * 2);      //  6 MB
  __bf16* wprojb  = (__bf16*)carve((size_t)Cc * Cc * 2);          //  2 MB
  float*  qkv_f   = (float*)carve((size_t)Mrows * 3 * Cc * 4);    // 48 MB
  __bf16* qb      = (__bf16*)carve((size_t)Bb * Hh * Nn * HD * 2);//  8 MB
  __bf16* kb      = (__bf16*)carve((size_t)Bb * Hh * Nn * HD * 2);//  8 MB
  __bf16* vT      = (__bf16*)carve((size_t)Bb * Hh * HD * Nn * 2);//  8 MB
  __bf16* attn_bf = (__bf16*)carve((size_t)Mrows * Cc * 2);       //  8 MB

  // 1) casts to bf16
  {
    int n4 = Mrows * Cc / 4;
    cast_f32_to_bf16<<<n4 / 256, 256, 0, stream>>>(x, xb, n4);
    n4 = 3 * Cc * Cc / 4;
    cast_f32_to_bf16<<<n4 / 256, 256, 0, stream>>>(qkv_w, wqkvb, n4);
    n4 = Cc * Cc / 4;
    cast_f32_to_bf16<<<n4 / 256, 256, 0, stream>>>(proj_w, wprojb, n4);
  }

  // 2) QKV GEMM: [4096,1024] x [3072,1024]^T -> [4096,3072] fp32
  {
    const int tiles = (Mrows / 32) * ((3 * Cc) / 64);  // 6144
    gemm_nt_bf16<<<tiles / 4, 128, 0, stream>>>(xb, wqkvb, qkv_f, nullptr,
                                                Mrows, 3 * Cc, Cc, tiles);
  }

  // 3) RMSNorm + RoPE + repack (one wave per (b,n,h))
  {
    const int waves = Bb * Nn * Hh;  // 65536
    prep_norm_rope<<<waves / 4, 128, 0, stream>>>(qkv_f, q_norm_w, k_norm_w,
                                                  pos_cos, pos_sin, qb, kb, vT);
  }

  // 4) causal flash attention (one wave per (b,h,qtile), 2 waves/block)
  {
    const int waves = Bb * Hh * (Nn / 16);  // 4096
    flash_attn<<<waves / 2, 64, 0, stream>>>(qb, kb, vT, attn_bf);
  }

  // 5) output projection: [4096,1024] x [1024,1024]^T + bias -> fp32 out
  {
    const int tiles = (Mrows / 32) * (Cc / 64);  // 2048
    gemm_nt_bf16<<<tiles / 4, 128, 0, stream>>>(attn_bf, wprojb, out, proj_b,
                                                Mrows, Cc, Cc, tiles);
  }
}
